// MultiLayerGCN_33217277067668
// MI455X (gfx1250) — compile-verified, matrix operand order checked by reading, not run on hardware
//
#include <hip/hip_runtime.h>

typedef _Float16 v16h __attribute__((ext_vector_type(16)));
typedef _Float16 v8h  __attribute__((ext_vector_type(8)));
typedef float    v8f  __attribute__((ext_vector_type(8)));

#define NNODES 100000
#define NEDGES 1600000
#define HDIM   128
#define ODIM   32

// ---------------- degree / norm ----------------

__global__ void k_init_deg(float* deg) {
    int i = blockIdx.x * blockDim.x + threadIdx.x;
    if (i < NNODES) deg[i] = 1.0f;           // self-loop contributes 1
}

__global__ void k_count_deg(const long long* ei, float* deg) {
    int e = blockIdx.x * blockDim.x + threadIdx.x;
    if (e < NEDGES) {
        int d = (int)ei[NEDGES + e];          // dst row
        atomicAdd(&deg[d], 1.0f);
    }
}

// dinv = rsqrt(deg); t = x*dinv; s1 init = self-loop term x[i]*dinv[i]^2
__global__ void k_dinv(const float* x, const float* deg,
                       float* dinv, float* t, float* s1) {
    int i = blockIdx.x * blockDim.x + threadIdx.x;
    if (i < NNODES) {
        float dg = deg[i];
        float di = dg > 0.0f ? rsqrtf(dg) : 0.0f;
        dinv[i] = di;
        float ti = x[i] * di;
        t[i] = ti;
        s1[i] = ti * di;                      // x[i] * dinv^2 (self loop)
    }
}

// ---------------- layer 1: scalar edge aggregation ----------------

__global__ void k_agg1(const long long* ei, const float* t,
                       const float* dinv, float* s1) {
    int e = blockIdx.x * blockDim.x + threadIdx.x;
    if (e < NEDGES) {
        int s = (int)ei[e];
        int d = (int)ei[NEDGES + e];
        atomicAdd(&s1[d], t[s] * dinv[d]);
    }
}

// h1[i,c] = relu(s1[i]*W1[c] + b1[c]) -> f16
__global__ void k_h1(const float* s1, const float* W1, const float* b1,
                     _Float16* h1h) {
    int idx = blockIdx.x * blockDim.x + threadIdx.x;
    if (idx < NNODES * HDIM) {
        int i = idx >> 7, c = idx & 127;
        float v = fmaf(s1[i], W1[c], b1[c]);
        h1h[idx] = (_Float16)(v > 0.0f ? v : 0.0f);
    }
}

// ---------------- weight transpose+convert (tiny) ----------------

__global__ void k_w2t(const float* W2, _Float16* W2T) {
    int idx = blockIdx.x * blockDim.x + threadIdx.x;   // 128*128
    if (idx < HDIM * HDIM) {
        int n = idx >> 7, k = idx & 127;
        W2T[n * HDIM + k] = (_Float16)W2[k * HDIM + n];
    }
}

__global__ void k_wfct(const float* Wfc, _Float16* WfcT) {
    int idx = blockIdx.x * blockDim.x + threadIdx.x;   // 32*128
    if (idx < ODIM * HDIM) {
        int n = idx / HDIM, k = idx % HDIM;
        WfcT[n * HDIM + k] = (_Float16)Wfc[k * ODIM + n];
    }
}

// ---------------- WMMA fragment helpers ----------------
// A-frag (16x32 f16): lane L holds row M=L&15, K = kb..kb+7 and kb+16..kb+23,
// kb = kk + (L<16 ? 0 : 8).  B-frag identical with lane = column, source
// pre-transposed to [n][k].  Both are two 16-byte contiguous loads.

__device__ __forceinline__ v16h load_frag_rowmajor(const _Float16* base,
                                                   int row, int kk, int lane) {
    int kb = kk + ((lane & 16) ? 8 : 0);
    const _Float16* p = base + row * HDIM + kb;
    v8h lo = *(const v8h*)(p);
    v8h hi = *(const v8h*)(p + 16);
    v16h r;
#pragma unroll
    for (int j = 0; j < 8; ++j) { r[j] = lo[j]; r[8 + j] = hi[j]; }
    return r;
}

// ---------------- GEMM1: H2pre = h1(f16) @ W2  (M=100000,K=128,N=128) -----
// Each wave computes a 16x64 strip: A fragment loaded once per K-slice and
// reused across 4 B tiles (16 WMMAs per wave).

__global__ void __launch_bounds__(32)
k_gemm1(const _Float16* h1h, const _Float16* W2T, float* H2) {
    int lane = threadIdx.x;
    int m0 = blockIdx.x * 16;
    int n0 = blockIdx.y * 64;
    int arow = m0 + (lane & 15);
    int bcol = lane & 15;
    v8f acc[4] = {};
#pragma unroll
    for (int kk = 0; kk < HDIM; kk += 32) {
        v16h a = load_frag_rowmajor(h1h, arow, kk, lane);
#pragma unroll
        for (int nt = 0; nt < 4; ++nt) {
            v16h b = load_frag_rowmajor(W2T, n0 + nt * 16 + bcol, kk, lane);
            acc[nt] = __builtin_amdgcn_wmma_f32_16x16x32_f16(
                false, a, false, b, (short)0, acc[nt], false, false);
        }
    }
    int mb = m0 + ((lane & 16) ? 8 : 0);
#pragma unroll
    for (int nt = 0; nt < 4; ++nt) {
        int n = n0 + nt * 16 + (lane & 15);
#pragma unroll
        for (int v = 0; v < 8; ++v)
            H2[(size_t)(mb + v) * HDIM + n] = acc[nt][v];
    }
}

// ---------------- layer 2 aggregation ----------------

// self-loop init: agg2[i,c] = H2[i,c] * dinv[i]^2
__global__ void k_agg2_init(const float* H2, const float* dinv, float* agg2) {
    int idx = blockIdx.x * blockDim.x + threadIdx.x;
    if (idx < NNODES * HDIM) {
        int i = idx >> 7;
        float di = dinv[i];
        agg2[idx] = H2[idx] * di * di;
    }
}

// one wave per edge: lane-uniform indices in SGPRs, each lane scatters a
// float4 channel group (coalesced 512B gather + f32 atomics into L2)
__global__ void k_agg2_edges(const long long* ei, const float* H2,
                             const float* dinv, float* agg2) {
    int wid = (blockIdx.x * blockDim.x + threadIdx.x) >> 5;   // edge id
    int lane = threadIdx.x & 31;
    if (wid < NEDGES) {
        int s = __builtin_amdgcn_readfirstlane((int)ei[wid]);
        int d = __builtin_amdgcn_readfirstlane((int)ei[NEDGES + wid]);
        float w = dinv[s] * dinv[d];
        int c = lane * 4;
        const float4 v = *(const float4*)(H2 + (size_t)s * HDIM + c);
        float* o = agg2 + (size_t)d * HDIM + c;
        atomicAdd(o + 0, v.x * w);
        atomicAdd(o + 1, v.y * w);
        atomicAdd(o + 2, v.z * w);
        atomicAdd(o + 3, v.w * w);
    }
}

// h2[i,c] = agg2[i,c] + b2[c]  -> f16  (no relu on last conv)
__global__ void k_h2(const float* agg2, const float* b2, _Float16* h2h) {
    int idx = blockIdx.x * blockDim.x + threadIdx.x;
    if (idx < NNODES * HDIM) {
        int c = idx & 127;
        h2h[idx] = (_Float16)(agg2[idx] + b2[c]);
    }
}

// ---------------- GEMM2: out = h2(f16) @ W_fc + b_fc  (N=32) ----------
// Each wave computes the full 16x32 output strip (8 WMMAs).

__global__ void __launch_bounds__(32)
k_gemm2(const _Float16* h2h, const _Float16* WfcT, const float* bfc,
        float* out) {
    int lane = threadIdx.x;
    int m0 = blockIdx.x * 16;
    int arow = m0 + (lane & 15);
    int bcol = lane & 15;
    v8f acc[2] = {};
#pragma unroll
    for (int kk = 0; kk < HDIM; kk += 32) {
        v16h a = load_frag_rowmajor(h2h, arow, kk, lane);
#pragma unroll
        for (int nt = 0; nt < 2; ++nt) {
            v16h b = load_frag_rowmajor(WfcT, nt * 16 + bcol, kk, lane);
            acc[nt] = __builtin_amdgcn_wmma_f32_16x16x32_f16(
                false, a, false, b, (short)0, acc[nt], false, false);
        }
    }
    int mb = m0 + ((lane & 16) ? 8 : 0);
#pragma unroll
    for (int nt = 0; nt < 2; ++nt) {
        int n = nt * 16 + (lane & 15);
        float bias = bfc[n];
#pragma unroll
        for (int v = 0; v < 8; ++v)
            out[(size_t)(mb + v) * ODIM + n] = acc[nt][v] + bias;
    }
}

// ---------------- launcher ----------------

extern "C" void kernel_launch(void* const* d_in, const int* in_sizes, int n_in,
                              void* d_out, int out_size, void* d_ws, size_t ws_size,
                              hipStream_t stream) {
    const float*     x   = (const float*)d_in[0];
    const long long* ei  = (const long long*)d_in[1];
    const float*     W1  = (const float*)d_in[2];
    const float*     b1  = (const float*)d_in[3];
    const float*     W2  = (const float*)d_in[4];
    const float*     b2  = (const float*)d_in[5];
    const float*     Wfc = (const float*)d_in[6];
    const float*     bfc = (const float*)d_in[7];
    float* out = (float*)d_out;

    char* w = (char*)d_ws;
    float* deg   = (float*)w;      w += (size_t)NNODES * 4;
    float* dinv  = (float*)w;      w += (size_t)NNODES * 4;
    float* t     = (float*)w;      w += (size_t)NNODES * 4;
    float* s1    = (float*)w;      w += (size_t)NNODES * 4;
    _Float16* h1h = (_Float16*)w;  w += (size_t)NNODES * HDIM * 2;
    _Float16* W2T = (_Float16*)w;  w += (size_t)HDIM * HDIM * 2;
    float* H2    = (float*)w;      w += (size_t)NNODES * HDIM * 4;
    float* agg2  = (float*)w;      w += (size_t)NNODES * HDIM * 4;
    _Float16* h2h = (_Float16*)w;  w += (size_t)NNODES * HDIM * 2;
    _Float16* WfcT = (_Float16*)w; w += (size_t)ODIM * HDIM * 2;

    const int B = 256;
    int gN   = (NNODES + B - 1) / B;
    int gE   = (NEDGES + B - 1) / B;
    int gNH  = (NNODES * HDIM + B - 1) / B;

    // degrees + norms
    k_init_deg<<<gN, B, 0, stream>>>(deg);
    k_count_deg<<<gE, B, 0, stream>>>(ei, deg);
    k_dinv<<<gN, B, 0, stream>>>(x, deg, dinv, t, s1);

    // layer 1 (scalar aggregation, then broadcast through W1 + relu)
    k_agg1<<<gE, B, 0, stream>>>(ei, t, dinv, s1);
    k_h1<<<gNH, B, 0, stream>>>(s1, W1, b1, h1h);

    // weights to f16 transposed
    k_w2t<<<(HDIM * HDIM + B - 1) / B, B, 0, stream>>>(W2, W2T);
    k_wfct<<<(ODIM * HDIM + B - 1) / B, B, 0, stream>>>(Wfc, WfcT);

    // GEMM1: 6250 x 2 strips of 16x64, one wave each (16 WMMAs/wave)
    k_gemm1<<<dim3(NNODES / 16, HDIM / 64), 32, 0, stream>>>(h1h, W2T, H2);

    // layer 2 aggregation: one wave per edge
    k_agg2_init<<<gNH, B, 0, stream>>>(H2, dinv, agg2);
    long long tot = (long long)NEDGES * 32;
    int gE32 = (int)((tot + B - 1) / B);
    k_agg2_edges<<<gE32, B, 0, stream>>>(ei, H2, dinv, agg2);
    k_h2<<<gNH, B, 0, stream>>>(agg2, b2, h2h);

    // GEMM2: 6250 strips of 16x32, fused bias (8 WMMAs/wave)
    k_gemm2<<<NNODES / 16, 32, 0, stream>>>(h2h, WfcT, bfc, out);
}
